// SemlaLayer_73607149519563
// MI455X (gfx1250) — matrix operand order repaired and checked by reading, not compiled, not run on hardware
//
#include <hip/hip_runtime.h>
#include <hip/hip_bf16.h>
#include <math.h>

// ---------------------------------------------------------------------------
// SEMLA layer for gfx1250 (MI455X).  B=4, N=256, D_EQUI=64, D_INV=256,
// D_MSG=64, N_HEADS=8, D_FF=256, IN_FEATS=256, D_OUT=136.
// Big pairwise MLP runs on v_wmma_f32_16x16x32_bf16.
// ---------------------------------------------------------------------------

typedef __bf16 bf16x8_t  __attribute__((ext_vector_type(8)));
typedef __bf16 v16bf_t   __attribute__((ext_vector_type(16)));
typedef float  v8f_t     __attribute__((ext_vector_type(8)));

union BF16x16 { v16bf_t v; bf16x8_t h[2]; };

#define NB   4
#define NN   256
#define DEQ  64
#define DINV 256
#define DMSG 64
#define DFF  256
#define DOUT 136
#define DOUTP 144          // D_OUT padded to multiple of 16
#define NEGINF (-1.0e9f)

// LDS tile: 64 rows x 256 cols bf16, padded row stride 264 (528B = +4 banks/row)
#define XSTRIDE 264

// ---------------- workspace layout (bytes) ----------------
#define WS_QM    0           // bf16  B*N*64          = 131072 B
#define WS_KM    131072      // bf16  B*N*64          = 131072 B
#define WS_W1BF  262144      // bf16  256*256         = 131072 B
#define WS_W2BF  393216      // bf16  144*256         =  73728 B
#define WS_PF    466944      // f32   B*N*256         = 1048576 B
#define WS_PE    1515520     // f32   B*N*3*64        =  786432 B
#define WS_EMSG  2301952     // f32   B*N*N*64        = 67108864 B
#define WS_IMSG  69410816    // f32   B*N*N*8         = 8388608 B
// total required: 77,799,424 bytes

// ---------------- output layout (floats) ----------------
#define OUT_EQ   0           // B*N*3*64  = 196608
#define OUT_INV  196608      // B*N*256   = 262144
#define OUT_EDGE 458752      // B*N*N*64  = 16777216

// ===========================================================================
// Prep 1: q_msg / k_msg (bf16) and proj_feats = invs@Wi^T+bi (f32)
// grid = B*N blocks of 128 threads
// ===========================================================================
__global__ __launch_bounds__(128) void semla_prep_proj(
    const float* __restrict__ invs,
    const float* __restrict__ Wq, const float* __restrict__ bq_,
    const float* __restrict__ Wk, const float* __restrict__ bk_,
    const float* __restrict__ Wi, const float* __restrict__ bi_,
    __bf16* __restrict__ qm, __bf16* __restrict__ km, float* __restrict__ pf)
{
    __shared__ float rowv[DINV];
    const int rn = blockIdx.x;                 // b*N + n
    for (int i = threadIdx.x; i < DINV; i += 128) rowv[i] = invs[rn * DINV + i];
    __syncthreads();
    for (int j = threadIdx.x; j < 384; j += 128) {
        const float* W;
        float bias;
        if (j < 64)       { W = Wq + j * DINV;         bias = bq_[j]; }
        else if (j < 128) { W = Wk + (j - 64) * DINV;  bias = bk_[j - 64]; }
        else              { W = Wi + (j - 128) * DINV; bias = bi_[j - 128]; }
        float acc = bias;
        for (int i = 0; i < DINV; ++i) acc += rowv[i] * W[i];
        if (j < 64)       qm[rn * DMSG + j]          = (__bf16)acc;
        else if (j < 128) km[rn * DMSG + (j - 64)]   = (__bf16)acc;
        else              pf[rn * DINV + (j - 128)]  = acc;
    }
}

// ===========================================================================
// Prep 2: proj_equi = equis@Wc^T.  grid = B*N*3 blocks of 64 threads
// ===========================================================================
__global__ __launch_bounds__(64) void semla_prep_equi(
    const float* __restrict__ equis, const float* __restrict__ Wc,
    float* __restrict__ pe)
{
    __shared__ float rowv[DEQ];
    const int r = blockIdx.x;                  // (b*N + n)*3 + c
    rowv[threadIdx.x] = equis[r * DEQ + threadIdx.x];
    __syncthreads();
    const int dout = threadIdx.x;
    float acc = 0.f;
    for (int d = 0; d < DEQ; ++d) acc += rowv[d] * Wc[dout * DEQ + d];
    pe[r * DEQ + dout] = acc;
}

// ===========================================================================
// Prep 3: bf16 weight copies (W2 zero-padded to 144 rows)
// grid = (65536+36864)/256 = 400 blocks of 256
// ===========================================================================
__global__ __launch_bounds__(256) void semla_prep_w(
    const float* __restrict__ W1, const float* __restrict__ W2,
    __bf16* __restrict__ W1bf, __bf16* __restrict__ W2bf)
{
    const int idx = blockIdx.x * 256 + threadIdx.x;
    if (idx < DFF * DFF) W1bf[idx] = (__bf16)W1[idx];
    const int j = idx - DFF * DFF;
    if (j >= 0 && j < DOUTP * DFF) {
        const int r = j >> 8, c = j & 255;
        W2bf[j] = (r < DOUT) ? (__bf16)W2[r * DFF + c] : (__bf16)0.f;
    }
}

// ===========================================================================
// Pairwise MLP (WMMA core).
// grid = B * (N/4) * (N/16) = 4096 blocks, 256 threads (8 waves).
// Block handles 64 pairwise rows: m = ql*16+kl, ql in 0..3, kl in 0..15.
// GEMM1: 64x256 @ 256x256 -> silu -> GEMM2: 64x256 @ 256x144.
// ===========================================================================
__global__ __launch_bounds__(256) void semla_pairwise_mlp(
    const float* __restrict__ equis,  const float* __restrict__ edges,
    const float* __restrict__ b1,     const float* __restrict__ b2,
    const __bf16* __restrict__ qm,    const __bf16* __restrict__ km,
    const __bf16* __restrict__ W1bf,  const __bf16* __restrict__ W2bf,
    float* __restrict__ emsg, float* __restrict__ imsg,
    float* __restrict__ edge_out)
{
    __shared__ __attribute__((aligned(16))) __bf16 tile[64 * XSTRIDE];

    const int tid  = threadIdx.x;
    const int lane = tid & 31;
    const int wave = tid >> 5;
    const int l16  = lane & 15;
    const int hi   = lane >> 4;                // 0 / 1

    const int bi = blockIdx.x;
    const int kt = bi & 15;                    // k tile (16 k's)
    const int qt = (bi >> 4) & 63;             // q tile (4 q's)
    const int b  = bi >> 10;

    // ---- build X: [q_msg | k_msg | dotprods | edges] in bf16 --------------
    {
        const int f = tid;                     // 0..255 (col)
        for (int m = 0; m < 64; ++m) {
            const int ql = m >> 4, kl = m & 15;
            const int q = qt * 4 + ql, k = kt * 16 + kl;
            __bf16 hv;
            if (f < 64) {
                hv = qm[(b * NN + q) * DMSG + f];
            } else if (f < 128) {
                hv = km[(b * NN + k) * DMSG + (f - 64)];
            } else if (f < 192) {
                const int d = f - 128;
                const float* eq = equis + (size_t)((b * NN + q) * 3) * DEQ + d;
                const float* ek = equis + (size_t)((b * NN + k) * 3) * DEQ + d;
                hv = (__bf16)(eq[0] * ek[0] + eq[64] * ek[64] + eq[128] * ek[128]);
            } else {
                hv = (__bf16)edges[((size_t)(b * NN + q) * NN + k) * DEQ + (f - 192)];
            }
            tile[m * XSTRIDE + f] = hv;
        }
    }
    __syncthreads();

    // wave tiling: mt = M tile (16 rows), nh selects N half (GEMM1) / N group
    const int mt = wave & 3;
    const int nh = wave >> 2;                  // 0 / 1
    const int arow = mt * 16 + l16;            // A-frag row for this lane

    // ---- GEMM1: acc[j] over N tiles nt = nh*8 + j ------------------------
    v8f_t acc[8];
#pragma unroll
    for (int j = 0; j < 8; ++j) acc[j] = 0.f;

    for (int kk = 0; kk < 8; ++kk) {           // K chunks of 32
        BF16x16 a;
        const __bf16* ap = &tile[arow * XSTRIDE + kk * 32 + hi * 8];
        a.h[0] = *(const bf16x8_t*)ap;         // K 0..7   (or 8..15)
        a.h[1] = *(const bf16x8_t*)(ap + 16);  // K 16..23 (or 24..31)
#pragma unroll
        for (int j = 0; j < 8; ++j) {
            const int n = (nh * 8 + j) * 16 + l16;          // output col / W1 row
            BF16x16 bb;
            const __bf16* bp = W1bf + n * DFF + kk * 32 + hi * 8;
            bb.h[0] = *(const bf16x8_t*)bp;
            bb.h[1] = *(const bf16x8_t*)(bp + 16);
            acc[j] = __builtin_amdgcn_wmma_f32_16x16x32_bf16(
                false, a.v, false, bb.v, (short)0, acc[j], false, false);
        }
    }
    __syncthreads();   // all waves done reading X

    // ---- bias + silu, write h (bf16) back into the same LDS tile ---------
#pragma unroll
    for (int j = 0; j < 8; ++j) {
        const int n = (nh * 8 + j) * 16 + l16;
        const float bn = b1[n];
#pragma unroll
        for (int r = 0; r < 8; ++r) {
            float v = acc[j][r] + bn;
            v = v / (1.f + __expf(-v));        // silu
            const int row = mt * 16 + r + hi * 8;
            tile[row * XSTRIDE + n] = (__bf16)v;
        }
    }
    __syncthreads();

    // ---- GEMM2: messages = h @ W2^T (N = 144, split 5/4 over nh) ----------
    const int ntBase  = (nh == 0) ? 0 : 5;
    const int ntCount = (nh == 0) ? 5 : 4;
    v8f_t acc2[5];
#pragma unroll
    for (int j = 0; j < 5; ++j) acc2[j] = 0.f;

    for (int kk = 0; kk < 8; ++kk) {
        BF16x16 a;
        const __bf16* ap = &tile[arow * XSTRIDE + kk * 32 + hi * 8];
        a.h[0] = *(const bf16x8_t*)ap;
        a.h[1] = *(const bf16x8_t*)(ap + 16);
#pragma unroll
        for (int j = 0; j < 5; ++j) {
            if (j < ntCount) {                 // wave-uniform predicate
                const int n = (ntBase + j) * 16 + l16;      // 0..143
                BF16x16 bb;
                const __bf16* bp = W2bf + n * DFF + kk * 32 + hi * 8;
                bb.h[0] = *(const bf16x8_t*)bp;
                bb.h[1] = *(const bf16x8_t*)(bp + 16);
                acc2[j] = __builtin_amdgcn_wmma_f32_16x16x32_bf16(
                    false, a.v, false, bb.v, (short)0, acc2[j], false, false);
            }
        }
    }

    // ---- bias + scatter: equi_msg / inv_msg / edge_out --------------------
#pragma unroll
    for (int j = 0; j < 5; ++j) {
        if (j < ntCount) {
            const int n = (ntBase + j) * 16 + l16;
            const float bn = (n < DOUT) ? b2[n] : 0.f;
#pragma unroll
            for (int r = 0; r < 8; ++r) {
                const int m  = mt * 16 + r + hi * 8;
                const int ql = m >> 4, kl = m & 15;
                const int q  = qt * 4 + ql, k = kt * 16 + kl;
                const float v = acc2[j][r] + bn;
                const int base = (b * NN + q) * NN + k;
                if (n < 64)       emsg[base * 64 + n] = v;
                else if (n < 72)  imsg[base * 8  + (n - 64)] = v;
                else if (n < DOUT) edge_out[base * 64 + (n - 72)] = v;
            }
        }
    }
}

// ===========================================================================
// Equi attention: per (b,q) block, 64 threads (one per channel d).
// 2-pass masked softmax fused with weighted sums over proj_equi, then Wa.
// ===========================================================================
__global__ __launch_bounds__(64) void semla_attn_equi(
    const float* __restrict__ emsg, const int* __restrict__ adj,
    const float* __restrict__ pe,   const float* __restrict__ Wa,
    float* __restrict__ out_eq)
{
    __shared__ float aout[3 * DEQ];
    const int bq = blockIdx.x;                 // b*N + q
    const int b  = bq >> 8;
    const int d  = threadIdx.x;

    const float* row  = emsg + (size_t)bq * NN * DEQ + d;
    const int*   arow = adj + bq * NN;

    float mx = -3.4e38f;
    for (int k = 0; k < NN; ++k) {
        const float x = row[(size_t)k * DEQ] + (arow[k] > 0 ? 0.f : NEGINF);
        mx = fmaxf(mx, x);
    }
    float s = 0.f, ss = 0.f, n0 = 0.f, n1 = 0.f, n2 = 0.f;
    const float* pbase = pe + (size_t)b * NN * 3 * DEQ + d;
    for (int k = 0; k < NN; ++k) {
        const float x = row[(size_t)k * DEQ] + (arow[k] > 0 ? 0.f : NEGINF);
        const float e = __expf(x - mx);
        s += e; ss += e * e;
        const float* p = pbase + (size_t)k * 3 * DEQ;
        n0 += e * p[0];
        n1 += e * p[64];
        n2 += e * p[128];
    }
    const float inv_s = 1.f / s;
    const float w = sqrtf(ss) * inv_s;         // w_e[d]
    aout[0 * DEQ + d] = n0 * inv_s * w;
    aout[1 * DEQ + d] = n1 * inv_s * w;
    aout[2 * DEQ + d] = n2 * inv_s * w;
    __syncthreads();

    for (int c = 0; c < 3; ++c) {              // equi_updates = aout @ Wa^T
        float acc = 0.f;
        for (int dd = 0; dd < DEQ; ++dd) acc += aout[c * DEQ + dd] * Wa[d * DEQ + dd];
        out_eq[(bq * 3 + c) * DEQ + d] = acc;
    }
}

// ===========================================================================
// Inv attention: per (b,q) block, 256 threads; thread t = (head h=t/32, d2=t%32).
// Redundant per-head softmax across the 32 threads of a head (cheap, no reduce).
// ===========================================================================
__global__ __launch_bounds__(256) void semla_attn_inv(
    const float* __restrict__ imsg, const int* __restrict__ adj,
    const float* __restrict__ pf,   const float* __restrict__ Wo,
    const float* __restrict__ bo,   float* __restrict__ out_inv)
{
    __shared__ float flat[DINV];
    const int bq = blockIdx.x;
    const int b  = bq >> 8;
    const int t  = threadIdx.x;
    const int h  = t >> 5;

    const float* row  = imsg + (size_t)bq * NN * 8 + h;
    const int*   arow = adj + bq * NN;

    float mx = -3.4e38f;
    for (int k = 0; k < NN; ++k) {
        const float x = row[(size_t)k * 8] + (arow[k] > 0 ? 0.f : NEGINF);
        mx = fmaxf(mx, x);
    }
    float s = 0.f, ss = 0.f, num = 0.f;
    const float* p = pf + (size_t)b * NN * DINV + t;   // head_feats[b,k,h,d2] = pf[b,k,t]
    for (int k = 0; k < NN; ++k) {
        const float x = row[(size_t)k * 8] + (arow[k] > 0 ? 0.f : NEGINF);
        const float e = __expf(x - mx);
        s += e; ss += e * e;
        num += e * p[(size_t)k * DINV];
    }
    const float inv_s = 1.f / s;
    flat[t] = num * inv_s * (sqrtf(ss) * inv_s);
    __syncthreads();

    float acc = bo[t];                         // inv_updates = flat @ Wo^T + bo
    for (int dd = 0; dd < DINV; ++dd) acc += flat[dd] * Wo[t * DINV + dd];
    out_inv[bq * DINV + t] = acc;
}

// ===========================================================================
extern "C" void kernel_launch(void* const* d_in, const int* in_sizes, int n_in,
                              void* d_out, int out_size, void* d_ws, size_t ws_size,
                              hipStream_t stream) {
    const float* equis = (const float*)d_in[0];
    const float* invs  = (const float*)d_in[1];
    const float* edges = (const float*)d_in[2];
    const int*   adj   = (const int*)  d_in[3];
    const float* Wq    = (const float*)d_in[4];
    const float* bq_   = (const float*)d_in[5];
    const float* Wk    = (const float*)d_in[6];
    const float* bk_   = (const float*)d_in[7];
    const float* W1    = (const float*)d_in[8];
    const float* b1    = (const float*)d_in[9];
    const float* W2    = (const float*)d_in[10];
    const float* b2    = (const float*)d_in[11];
    const float* Wc    = (const float*)d_in[12];
    const float* Wa    = (const float*)d_in[13];
    const float* Wi    = (const float*)d_in[14];
    const float* bi_   = (const float*)d_in[15];
    const float* Wo    = (const float*)d_in[16];
    const float* bo    = (const float*)d_in[17];

    char* ws = (char*)d_ws;                    // requires ~78 MB of scratch
    __bf16* qm    = (__bf16*)(ws + WS_QM);
    __bf16* km    = (__bf16*)(ws + WS_KM);
    __bf16* W1bf  = (__bf16*)(ws + WS_W1BF);
    __bf16* W2bf  = (__bf16*)(ws + WS_W2BF);
    float*  pf    = (float*)(ws + WS_PF);
    float*  pe    = (float*)(ws + WS_PE);
    float*  emsg  = (float*)(ws + WS_EMSG);
    float*  imsg  = (float*)(ws + WS_IMSG);

    float* out    = (float*)d_out;
    float* out_eq   = out + OUT_EQ;
    float* out_inv  = out + OUT_INV;
    float* out_edge = out + OUT_EDGE;

    semla_prep_proj<<<NB * NN, 128, 0, stream>>>(invs, Wq, bq_, Wk, bk_, Wi, bi_,
                                                 qm, km, pf);
    semla_prep_equi<<<NB * NN * 3, 64, 0, stream>>>(equis, Wc, pe);
    semla_prep_w<<<(DFF * DFF + DOUTP * DFF + 255) / 256, 256, 0, stream>>>(
        W1, W2, W1bf, W2bf);

    semla_pairwise_mlp<<<NB * (NN / 4) * (NN / 16), 256, 0, stream>>>(
        equis, edges, b1, b2, qm, km, W1bf, W2bf, emsg, imsg, out_edge);

    semla_attn_equi<<<NB * NN, 64, 0, stream>>>(emsg, adj, pe, Wa, out_eq);
    semla_attn_inv<<<NB * NN, 256, 0, stream>>>(imsg, adj, pf, Wo, bo, out_inv);
}
